// DecoderRNN_52080773431977
// MI455X (gfx1250) — matrix operand order repaired and checked by reading, not compiled, not run on hardware
//
#include <hip/hip_runtime.h>
#include <hip/hip_bf16.h>

// ---------------- problem constants ----------------
#define B_    128
#define P_    196
#define T_    20
#define VOCAB_ 10000
#define ENC_  512
#define EMB_  512
#define DEC_  512
#define ATTN_ 512

typedef __attribute__((ext_vector_type(16))) __bf16 v16bf;
typedef __attribute__((ext_vector_type(8)))  float  v8f;

__device__ __forceinline__ unsigned short f2bf(float f) {
    unsigned int u = __float_as_uint(f);
    // round-to-nearest-even fp32 -> bf16
    unsigned int r = (u + 0x7FFFu + ((u >> 16) & 1u)) >> 16;
    return (unsigned short)r;
}

union Frag { v16bf v; uint4 q[2]; };

// A fragment: 16x32 bf16 tile row for this lane; two b128s 32 bytes apart
__device__ __forceinline__ void loadA(Frag& f, const unsigned short* p) {
    f.q[0] = *(const uint4*)(p);
    f.q[1] = *(const uint4*)(p + 16);
}
// B fragment: column n of rows k = 8h..8h+7 and 16+8h..16+8h+7 (stride ld)
__device__ __forceinline__ void loadB(Frag& f, const unsigned short* p, int ld) {
    unsigned short* d = (unsigned short*)&f;
#pragma unroll
    for (int s = 0; s < 8; ++s) {
        d[s]     = p[(long long)s * ld];
        d[s + 8] = p[(long long)(s + 16) * ld];
    }
}

// =====================================================================
// GEMM: C[M,N] = A_bf16[M,K](lda) * Wt_bf16[K,N] + bias[N]
// A row-major bf16; Wt transposed (K-major) bf16.
// Per-wave 64x16 C tile: 4 A fragments share 1 B fragment -> 4 WMMAs/K-step.
// Software-pipelined (double-buffered) K loop: next step's 8 b128 A loads + 16 B element loads
// are in flight while the current 4 WMMAs execute.
// NT=true streams C through non-temporal stores (write-once outputs).
// Requires: M % 64 == 0, N % 16 == 0, K % 64 == 0 (K/32 steps, unroll 2).
// =====================================================================
template <bool NT>
__global__ void __launch_bounds__(256)
gemm_bf16_wmma(const unsigned short* __restrict__ A, int lda,
               const unsigned short* __restrict__ Wt,
               const float* __restrict__ bias,
               float* __restrict__ C, long long ldc,
               int M, int N, int K)
{
    int wave = (int)((blockIdx.x * blockDim.x + threadIdx.x) >> 5);
    int lane = (int)(threadIdx.x & 31);
    int Mt = M >> 6;   // 64-row tiles
    int Nt = N >> 4;   // 16-col tiles
    if (wave >= Mt * Nt) return;          // whole-wave uniform exit
    int tm = wave % Mt;
    int tn = wave / Mt;                   // consecutive waves share B column

    int half = lane >> 4;
    long long mbase = (long long)(tm * 64 + (lane & 15));
    int n0 = tn * 16;

    const unsigned short* ap0 = A + (mbase      ) * lda + half * 8;
    const unsigned short* ap1 = A + (mbase + 16 ) * lda + half * 8;
    const unsigned short* ap2 = A + (mbase + 32 ) * lda + half * 8;
    const unsigned short* ap3 = A + (mbase + 48 ) * lda + half * 8;
    const unsigned short* bp  = Wt + (long long)(half * 8) * N + n0 + (lane & 15);
    const long long bstep = (long long)32 * N;

    v8f acc0 = {}, acc1 = {}, acc2 = {}, acc3 = {};

    // ---------------- prologue: stage first K-step ----------------
    Frag a0, a1, a2, a3, b;
    loadA(a0, ap0); loadA(a1, ap1); loadA(a2, ap2); loadA(a3, ap3);
    loadB(b, bp, N);

    int nsteps = K >> 5;
#pragma unroll 2
    for (int s = 1; s < nsteps; ++s) {
        ap0 += 32; ap1 += 32; ap2 += 32; ap3 += 32; bp += bstep;

        // issue next step's loads before consuming current fragments
        Frag na0, na1, na2, na3, nb;
        loadA(na0, ap0); loadA(na1, ap1); loadA(na2, ap2); loadA(na3, ap3);
        loadB(nb, bp, N);
        __builtin_prefetch(bp + bstep, 0, 1);   // speculative next B row

        acc0 = __builtin_amdgcn_wmma_f32_16x16x32_bf16(false, a0.v, false, b.v,
                                                       (short)0, acc0, false, false);
        acc1 = __builtin_amdgcn_wmma_f32_16x16x32_bf16(false, a1.v, false, b.v,
                                                       (short)0, acc1, false, false);
        acc2 = __builtin_amdgcn_wmma_f32_16x16x32_bf16(false, a2.v, false, b.v,
                                                       (short)0, acc2, false, false);
        acc3 = __builtin_amdgcn_wmma_f32_16x16x32_bf16(false, a3.v, false, b.v,
                                                       (short)0, acc3, false, false);

        a0 = na0; a1 = na1; a2 = na2; a3 = na3; b = nb;
    }
    // ---------------- epilogue: last K-step ----------------
    acc0 = __builtin_amdgcn_wmma_f32_16x16x32_bf16(false, a0.v, false, b.v,
                                                   (short)0, acc0, false, false);
    acc1 = __builtin_amdgcn_wmma_f32_16x16x32_bf16(false, a1.v, false, b.v,
                                                   (short)0, acc1, false, false);
    acc2 = __builtin_amdgcn_wmma_f32_16x16x32_bf16(false, a2.v, false, b.v,
                                                   (short)0, acc2, false, false);
    acc3 = __builtin_amdgcn_wmma_f32_16x16x32_bf16(false, a3.v, false, b.v,
                                                   (short)0, acc3, false, false);

    int nc = n0 + (lane & 15);
    float bs = bias ? bias[nc] : 0.0f;
    long long mout = (long long)(tm * 64 + half * 8);
    float* c0 = C + mout * ldc + nc;
#pragma unroll
    for (int r = 0; r < 8; ++r) {
        float v = acc0[r] + bs;
        if (NT) __builtin_nontemporal_store(v, c0 + (long long)r * ldc);
        else    c0[(long long)r * ldc] = v;
    }
    c0 += 16 * ldc;
#pragma unroll
    for (int r = 0; r < 8; ++r) {
        float v = acc1[r] + bs;
        if (NT) __builtin_nontemporal_store(v, c0 + (long long)r * ldc);
        else    c0[(long long)r * ldc] = v;
    }
    c0 += 16 * ldc;
#pragma unroll
    for (int r = 0; r < 8; ++r) {
        float v = acc2[r] + bs;
        if (NT) __builtin_nontemporal_store(v, c0 + (long long)r * ldc);
        else    c0[(long long)r * ldc] = v;
    }
    c0 += 16 * ldc;
#pragma unroll
    for (int r = 0; r < 8; ++r) {
        float v = acc3[r] + bs;
        if (NT) __builtin_nontemporal_store(v, c0 + (long long)r * ldc);
        else    c0[(long long)r * ldc] = v;
    }
}

// =====================================================================
// Weight convert + transpose:  Wt[(k+krow0)*ldn + ncol0 + n] = bf16(W[n*K+k])
// =====================================================================
__global__ void wconv_kernel(const float* __restrict__ W,
                             unsigned short* __restrict__ Wt,
                             int N, int K, int ldn, int ncol0, int krow0)
{
    long long idx = (long long)blockIdx.x * blockDim.x + threadIdx.x;
    if (idx >= (long long)N * K) return;
    int k = (int)(idx % K);
    int n = (int)(idx / K);
    Wt[(long long)(k + krow0) * ldn + ncol0 + n] = f2bf(W[(long long)n * K + k]);
}

// plain fp32 -> bf16 elementwise
__global__ void f2bf_kernel(const float* __restrict__ in,
                            unsigned short* __restrict__ out, long long n)
{
    long long idx = (long long)blockIdx.x * blockDim.x + threadIdx.x;
    if (idx < n) out[idx] = f2bf(in[idx]);
}

// extract h (cols 0..511 of hc[.,1024]) as bf16
__global__ void h2bf_kernel(const float* __restrict__ hc,
                            unsigned short* __restrict__ h_bf)
{
    int idx = blockIdx.x * blockDim.x + threadIdx.x;
    if (idx >= B_ * DEC_) return;
    int b = idx >> 9, e = idx & (DEC_ - 1);
    h_bf[idx] = f2bf(hc[(long long)b * 1024 + e]);
}

__global__ void bias_cat_kernel(const float* __restrict__ b_ih,
                                const float* __restrict__ b_hh,
                                float* __restrict__ out)
{
    int i = blockIdx.x * blockDim.x + threadIdx.x;
    if (i < 4 * DEC_) out[i] = b_ih[i] + b_hh[i];
}

__global__ void bias_init_kernel(const float* __restrict__ bh,
                                 const float* __restrict__ bc,
                                 float* __restrict__ out)
{
    int i = blockIdx.x * blockDim.x + threadIdx.x;
    if (i < 2 * DEC_) out[i] = (i < DEC_) ? bh[i] : bc[i - DEC_];
}

// mean over P of encoder_out -> [B, ENC] (bf16 for the init GEMM)
__global__ void mean_enc_kernel(const float* __restrict__ enc,
                                unsigned short* __restrict__ meanv)
{
    int idx = blockIdx.x * blockDim.x + threadIdx.x;
    if (idx >= B_ * ENC_) return;
    int b = idx >> 9, e = idx & (ENC_ - 1);
    const float* p = enc + (long long)b * P_ * ENC_ + e;
    float s = 0.f;
    for (int i = 0; i < P_; ++i) s += p[(long long)i * ENC_];
    meanv[idx] = f2bf(s * (1.0f / (float)P_));
}

// embedding gather (bf16): embs[b,t,:] = bf16(embedding[captions[b,t]])
__global__ void embed_kernel(const float* __restrict__ emb,
                             const int* __restrict__ cap,
                             unsigned short* __restrict__ out)
{
    long long idx = (long long)blockIdx.x * blockDim.x + threadIdx.x;
    if (idx >= (long long)B_ * T_ * EMB_) return;
    int e = (int)(idx & (EMB_ - 1));
    long long bt = idx >> 9;
    int tok = cap[bt];
    out[idx] = f2bf(emb[(long long)tok * EMB_ + e]);
}

// =====================================================================
// Fused additive-attention: energy -> softmax -> alpha write -> context
// one block per batch element, 256 threads (8 waves)
// =====================================================================
__global__ void attention_kernel(const float* __restrict__ attn_enc, // [B,P,ATTN]
                                 const float* __restrict__ attn_dec, // [B,ATTN]
                                 const float* __restrict__ full_w,   // [ATTN]
                                 const float* __restrict__ full_b,   // [1]
                                 const float* __restrict__ enc,      // [B,P,ENC]
                                 float* __restrict__ alphas,         // [B,T,P]
                                 float* __restrict__ context,        // [B,ENC]
                                 int t)
{
    int b    = blockIdx.x;
    int tid  = threadIdx.x;
    int lane = tid & 31;
    int wv   = tid >> 5;

    __shared__ float s_dec[ATTN_];
    __shared__ float s_w[ATTN_];
    __shared__ float s_e[P_];
    __shared__ float s_red[8];

    for (int i = tid; i < ATTN_; i += 256) {
        s_dec[i] = attn_dec[(long long)b * ATTN_ + i];
        s_w[i]   = full_w[i];
    }
    __syncthreads();

    float fb = full_b[0];
    const float* ae = attn_enc + (long long)b * P_ * ATTN_;

    // energies: one wave per p, lanes reduce over ATTN
    for (int p = wv; p < P_; p += 8) {
        const float* row = ae + (long long)p * ATTN_;
        float acc = 0.f;
        for (int a = lane; a < ATTN_; a += 32) {
            float v = row[a] + s_dec[a];
            acc += (v > 0.f ? v : 0.f) * s_w[a];
        }
        for (int off = 16; off > 0; off >>= 1) acc += __shfl_down(acc, off, 32);
        if (lane == 0) s_e[p] = acc + fb;
    }
    __syncthreads();

    // softmax max
    float m = -1e30f;
    for (int p = tid; p < P_; p += 256) m = fmaxf(m, s_e[p]);
    for (int off = 16; off > 0; off >>= 1) m = fmaxf(m, __shfl_down(m, off, 32));
    if (lane == 0) s_red[wv] = m;
    __syncthreads();
    if (tid == 0) {
        float mm = s_red[0];
        for (int i = 1; i < 8; ++i) mm = fmaxf(mm, s_red[i]);
        s_red[0] = mm;
    }
    __syncthreads();
    m = s_red[0];
    __syncthreads();

    // exp + sum
    float ssum = 0.f;
    for (int p = tid; p < P_; p += 256) {
        float e = __expf(s_e[p] - m);
        s_e[p]  = e;
        ssum   += e;
    }
    for (int off = 16; off > 0; off >>= 1) ssum += __shfl_down(ssum, off, 32);
    if (lane == 0) s_red[wv] = ssum;
    __syncthreads();
    if (tid == 0) {
        float s = 0.f;
        for (int i = 0; i < 8; ++i) s += s_red[i];
        s_red[0] = s;
    }
    __syncthreads();
    float inv = 1.0f / s_red[0];
    for (int p = tid; p < P_; p += 256) s_e[p] *= inv;
    __syncthreads();

    // write alphas
    for (int p = tid; p < P_; p += 256)
        alphas[((long long)b * T_ + t) * P_ + p] = s_e[p];

    // context[b,e] = sum_p alpha[p] * enc[b,p,e]
    const float* eb = enc + (long long)b * P_ * ENC_;
    for (int e = tid; e < ENC_; e += 256) {
        float acc = 0.f;
        for (int p = 0; p < P_; ++p) acc += s_e[p] * eb[(long long)p * ENC_ + e];
        context[(long long)b * ENC_ + e] = acc;
    }
}

// pack xcat_bf16 = [emb_t(512) | context(512) | h(512)]
__global__ void pack_x_kernel(const unsigned short* __restrict__ embs, int t,
                              const float* __restrict__ context,
                              const float* __restrict__ hc,
                              unsigned short* __restrict__ xcat)
{
    int idx = blockIdx.x * blockDim.x + threadIdx.x;
    if (idx >= B_ * 1536) return;
    int b = idx / 1536, j = idx % 1536;
    unsigned short v;
    if (j < 512)       v = embs[((long long)b * T_ + t) * EMB_ + j];
    else if (j < 1024) v = f2bf(context[(long long)b * ENC_ + (j - 512)]);
    else               v = f2bf(hc[(long long)b * 1024 + (j - 1024)]);
    xcat[idx] = v;
}

// LSTM cell elementwise; hc holds [h(512) | c(512)] per row; also emit h_bf16
__global__ void lstm_kernel(const float* __restrict__ gates,
                            float* __restrict__ hc,
                            unsigned short* __restrict__ h_bf)
{
    int idx = blockIdx.x * blockDim.x + threadIdx.x;
    if (idx >= B_ * DEC_) return;
    int b = idx >> 9, e = idx & (DEC_ - 1);
    const float* g = gates + (long long)b * 4 * DEC_;
    float gi = g[e], gf = g[DEC_ + e], gg = g[2 * DEC_ + e], go = g[3 * DEC_ + e];
    float c  = hc[(long long)b * 1024 + DEC_ + e];
    float si = 1.f / (1.f + __expf(-gi));
    float sf = 1.f / (1.f + __expf(-gf));
    float so = 1.f / (1.f + __expf(-go));
    float cn = sf * c + si * tanhf(gg);
    float hn = so * tanhf(cn);
    hc[(long long)b * 1024 + e]        = hn;
    hc[(long long)b * 1024 + DEC_ + e] = cn;
    h_bf[idx] = f2bf(hn);
}

// ---------------------------------------------------------------------
static inline void launch_gemm(const unsigned short* A, int lda,
                               const unsigned short* Wt,
                               const float* bias, float* C, long long ldc,
                               int M, int N, int K, bool nt, hipStream_t s)
{
    int waves  = (M / 64) * (N / 16);
    int blocks = (waves + 7) / 8;
    if (nt)
        gemm_bf16_wmma<true><<<blocks, 256, 0, s>>>(A, lda, Wt, bias, C, ldc, M, N, K);
    else
        gemm_bf16_wmma<false><<<blocks, 256, 0, s>>>(A, lda, Wt, bias, C, ldc, M, N, K);
}

static inline size_t alignup(size_t x) { return (x + 255) & ~(size_t)255; }

extern "C" void kernel_launch(void* const* d_in, const int* in_sizes, int n_in,
                              void* d_out, int out_size, void* d_ws, size_t ws_size,
                              hipStream_t stream)
{
    (void)in_sizes; (void)n_in; (void)out_size; (void)ws_size;
    const float* encoder_out = (const float*)d_in[0];
    const int*   captions    = (const int*)  d_in[1];
    /* lengths d_in[2] unused by the reference computation */
    const float* embedding   = (const float*)d_in[3];
    const float* enc_attn_w  = (const float*)d_in[4];
    const float* enc_attn_b  = (const float*)d_in[5];
    const float* dec_attn_w  = (const float*)d_in[6];
    const float* dec_attn_b  = (const float*)d_in[7];
    const float* full_attn_w = (const float*)d_in[8];
    const float* full_attn_b = (const float*)d_in[9];
    const float* W_ih        = (const float*)d_in[10];
    const float* b_ih        = (const float*)d_in[11];
    const float* W_hh        = (const float*)d_in[12];
    const float* b_hh        = (const float*)d_in[13];
    const float* init_h_w    = (const float*)d_in[14];
    const float* init_h_b    = (const float*)d_in[15];
    const float* init_c_w    = (const float*)d_in[16];
    const float* init_c_b    = (const float*)d_in[17];
    const float* fc_w        = (const float*)d_in[18];
    const float* fc_b        = (const float*)d_in[19];

    float* outputs = (float*)d_out;                                   // [B,T,VOCAB]
    float* alphas  = outputs + (long long)B_ * T_ * VOCAB_;           // [B,T,P]

    // ---- workspace carve-out (bytes) ----
    char* ws = (char*)d_ws;
    size_t cur = 0;
    auto carve = [&](size_t bytes) { char* p = ws + cur; cur += alignup(bytes); return p; };
    unsigned short* Wt_enc  = (unsigned short*)carve((size_t)ATTN_ * ENC_ * 2);        // [K=512,N=512]
    unsigned short* Wt_dec  = (unsigned short*)carve((size_t)ATTN_ * DEC_ * 2);
    unsigned short* Wt_init = (unsigned short*)carve((size_t)ENC_ * 1024 * 2);         // [512,1024]
    unsigned short* Wt_cat  = (unsigned short*)carve((size_t)1536 * 2048 * 2);         // [1536,2048]
    unsigned short* Wt_fc   = (unsigned short*)carve((size_t)DEC_ * VOCAB_ * 2);       // [512,10000]
    float* bias_cat  = (float*)carve(4 * DEC_ * 4);
    float* bias_init = (float*)carve(2 * DEC_ * 4);
    unsigned short* mean_bf = (unsigned short*)carve((size_t)B_ * ENC_ * 2);
    unsigned short* h_bf    = (unsigned short*)carve((size_t)B_ * DEC_ * 2);
    unsigned short* enc_bf  = (unsigned short*)carve((size_t)B_ * P_ * ENC_ * 2);      // encoder_out bf16
    unsigned short* embs    = (unsigned short*)carve((size_t)B_ * T_ * EMB_ * 2);
    unsigned short* xcat    = (unsigned short*)carve((size_t)B_ * 1536 * 2);
    float* hc        = (float*)carve((size_t)B_ * 1024 * 4);                           // [h|c] fp32
    float* attn_dec  = (float*)carve((size_t)B_ * ATTN_ * 4);
    float* context   = (float*)carve((size_t)B_ * ENC_ * 4);
    float* gates     = (float*)carve((size_t)B_ * 4 * DEC_ * 4);
    float* attn_enc  = (float*)carve((size_t)B_ * P_ * ATTN_ * 4);

    // ================= prep (each call; deterministic) =================
    {   // weight transpose + bf16 conversion
        long long n;
        n = (long long)ATTN_ * ENC_;
        wconv_kernel<<<(unsigned)((n + 255) / 256), 256, 0, stream>>>(enc_attn_w, Wt_enc, ATTN_, ENC_, ATTN_, 0, 0);
        wconv_kernel<<<(unsigned)((n + 255) / 256), 256, 0, stream>>>(dec_attn_w, Wt_dec, ATTN_, DEC_, ATTN_, 0, 0);
        wconv_kernel<<<(unsigned)((n + 255) / 256), 256, 0, stream>>>(init_h_w, Wt_init, DEC_, ENC_, 1024, 0, 0);
        wconv_kernel<<<(unsigned)((n + 255) / 256), 256, 0, stream>>>(init_c_w, Wt_init, DEC_, ENC_, 1024, 512, 0);
        n = (long long)2048 * 1024;
        wconv_kernel<<<(unsigned)((n + 255) / 256), 256, 0, stream>>>(W_ih, Wt_cat, 2048, 1024, 2048, 0, 0);
        n = (long long)2048 * 512;
        wconv_kernel<<<(unsigned)((n + 255) / 256), 256, 0, stream>>>(W_hh, Wt_cat, 2048, 512, 2048, 0, 1024);
        n = (long long)VOCAB_ * DEC_;
        wconv_kernel<<<(unsigned)((n + 255) / 256), 256, 0, stream>>>(fc_w, Wt_fc, VOCAB_, DEC_, VOCAB_, 0, 0);
    }
    bias_cat_kernel<<<(4 * DEC_ + 255) / 256, 256, 0, stream>>>(b_ih, b_hh, bias_cat);
    bias_init_kernel<<<(2 * DEC_ + 255) / 256, 256, 0, stream>>>(init_h_b, init_c_b, bias_init);
    mean_enc_kernel<<<(B_ * ENC_ + 255) / 256, 256, 0, stream>>>(encoder_out, mean_bf);
    {
        long long n = (long long)B_ * P_ * ENC_;   // encoder_out -> bf16 (A of attn_enc GEMM)
        f2bf_kernel<<<(unsigned)((n + 255) / 256), 256, 0, stream>>>(encoder_out, enc_bf, n);
        n = (long long)B_ * T_ * EMB_;
        embed_kernel<<<(unsigned)((n + 255) / 256), 256, 0, stream>>>(embedding, captions, embs);
    }
    // hc = mean_enc @ [init_h_w|init_c_w]^T + [b_h|b_c]   -> [B,1024] fp32
    launch_gemm(mean_bf, ENC_, Wt_init, bias_init, hc, 1024, B_, 1024, ENC_, false, stream);
    h2bf_kernel<<<(B_ * DEC_ + 255) / 256, 256, 0, stream>>>(hc, h_bf);
    // attn_enc = encoder_out @ enc_attn_w^T + b           -> [B*P, ATTN] fp32
    launch_gemm(enc_bf, ENC_, Wt_enc, enc_attn_b, attn_enc, ATTN_,
                B_ * P_, ATTN_, ENC_, false, stream);

    // ================= decode loop =================
    for (int t = 0; t < T_; ++t) {
        // attn_dec = h @ dec_attn_w^T + b
        launch_gemm(h_bf, DEC_, Wt_dec, dec_attn_b, attn_dec, ATTN_, B_, ATTN_, DEC_, false, stream);

        attention_kernel<<<B_, 256, 0, stream>>>(attn_enc, attn_dec, full_attn_w,
                                                 full_attn_b, encoder_out,
                                                 alphas, context, t);

        pack_x_kernel<<<(B_ * 1536 + 255) / 256, 256, 0, stream>>>(embs, t, context, hc, xcat);

        // gates = [x|h] @ [W_ih|W_hh]^T + (b_ih + b_hh)
        launch_gemm(xcat, 1536, Wt_cat, bias_cat, gates, 4 * DEC_, B_, 4 * DEC_, 1536, false, stream);

        lstm_kernel<<<(B_ * DEC_ + 255) / 256, 256, 0, stream>>>(gates, hc, h_bf);

        // outputs[:, t, :] = h_new @ fc_w^T + fc_b  (ldc = T*VOCAB, NT streamed)
        launch_gemm(h_bf, DEC_, Wt_fc, fc_b, outputs + (long long)t * VOCAB_,
                    (long long)T_ * VOCAB_, B_, VOCAB_, DEC_, true, stream);
    }
}
